// OuterProductMeanSM_44375602102639
// MI455X (gfx1250) — matrix operand order repaired and checked by reading, not compiled, not run on hardware
//
#include <hip/hip_runtime.h>

typedef __attribute__((ext_vector_type(16))) _Float16 v16h;
typedef __attribute__((ext_vector_type(8)))  _Float16 v8h;
typedef __attribute__((ext_vector_type(8)))  float    v8f;
typedef __attribute__((ext_vector_type(4)))  float    v4f;

#define L1   256
#define KDIM 128
#define L2   1024
#define CS   384
#define CM   256
#define CH   32
#define CZ   128
#define LN_EPS 1e-5f

// ---------------------------------------------------------------------------
// K1: a[i,e] = LN(s[i,:]) @ W_ab + b_ab      (256 rows x 384 -> 256 x 32)
// one block of 128 threads per row i
// ---------------------------------------------------------------------------
__global__ void k_ln_proj_s(const float* __restrict__ s,
                            const float* __restrict__ g,
                            const float* __restrict__ b,
                            const float* __restrict__ W,
                            const float* __restrict__ bias,
                            float* __restrict__ a)
{
    __shared__ float srow[CS];
    __shared__ float red[128];
    __shared__ float s_mu, s_rs;
    const int i = blockIdx.x;
    const int t = threadIdx.x;              // 0..127
    const float* row = s + (size_t)i * CS;

    float ps = 0.f, pq = 0.f;
    for (int c = t; c < CS; c += 128) {
        float x = row[c];
        srow[c] = x;
        ps += x; pq += x * x;
    }
    red[t] = ps; __syncthreads();
    for (int off = 64; off; off >>= 1) { if (t < off) red[t] += red[t + off]; __syncthreads(); }
    if (t == 0) s_mu = red[0] * (1.f / CS);
    __syncthreads();
    red[t] = pq; __syncthreads();
    for (int off = 64; off; off >>= 1) { if (t < off) red[t] += red[t + off]; __syncthreads(); }
    if (t == 0) {
        float mu  = s_mu;
        float var = red[0] * (1.f / CS) - mu * mu;
        s_rs = rsqrtf(var + LN_EPS);
    }
    __syncthreads();
    const float mu = s_mu, rs = s_rs;

    // projection: thread t -> (e = t%32, quarter q = t/32 covering 96 c's)
    const int e = t & 31, q = t >> 5;
    float acc = 0.f;
    for (int c = q * 96; c < q * 96 + 96; ++c) {
        float xn = (srow[c] - mu) * rs * g[c] + b[c];
        acc += xn * W[c * CH + e];
    }
    __syncthreads();
    red[t] = acc; __syncthreads();
    if (t < CH)
        a[i * CH + e] = red[e] + red[32 + e] + red[64 + e] + red[96 + e] + bias[e];
}

// ---------------------------------------------------------------------------
// K2: Tt_h[i][z][e] = f16( inv_norm * sum_c a[i,c] * W_out[(c*32+e)*128+z] )
// block (32 x 8): x = e, y = z-chunk ; grid (256, 16)
// ---------------------------------------------------------------------------
__global__ void k_make_T(const float* __restrict__ a,
                         const float* __restrict__ Wout,
                         _Float16* __restrict__ Tt,
                         float inv_norm)
{
    const int i = blockIdx.x;
    const int e = threadIdx.x;                                  // 0..31
    const int z = blockIdx.y * blockDim.y + threadIdx.y;        // 0..127
    const float* ai = a + i * CH;
    float acc = 0.f;
#pragma unroll
    for (int c = 0; c < CH; ++c)
        acc += ai[c] * Wout[(c * CH + e) * CZ + z];
    Tt[((size_t)i * CZ + z) * CH + e] = (_Float16)(acc * inv_norm);
}

// ---------------------------------------------------------------------------
// K3: msum_h[j,e] = f16( sum_c (sum_k LN(m[k,j,:]))[c] * W_ag[c,e] + K*b_ag[e] )
// one block (256 threads = 8 waves) per j; wave w handles k = w, w+8, ...
// lane owns c = lane*8 .. lane*8+7 (fixed across k -> register accumulation)
// ---------------------------------------------------------------------------
__global__ void k_msum(const float* __restrict__ m,
                       const float* __restrict__ g,
                       const float* __restrict__ b,
                       const float* __restrict__ Wag,
                       const float* __restrict__ bag,
                       _Float16* __restrict__ msum_h)
{
    __shared__ float Ssh[8][CM];
    const int j    = blockIdx.x;
    const int t    = threadIdx.x;       // 0..255
    const int w    = t >> 5;
    const int lane = t & 31;
    const int c0   = lane * 8;

    float gv[8], bv[8], accS[8];
#pragma unroll
    for (int u = 0; u < 8; ++u) { gv[u] = g[c0 + u]; bv[u] = b[c0 + u]; accS[u] = 0.f; }

    for (int it = 0; it < 16; ++it) {
        const int k = w + 8 * it;
        const float* row = m + ((size_t)(k * L2 + j)) * CM;
        v4f x0 = *(const v4f*)(row + c0);
        v4f x1 = *(const v4f*)(row + c0 + 4);
        float ps = 0.f, pq = 0.f;
#pragma unroll
        for (int u = 0; u < 8; ++u) {
            float xv = (u < 4) ? x0[u] : x1[u - 4];
            ps += xv; pq += xv * xv;
        }
#pragma unroll
        for (int off = 16; off; off >>= 1) {          // wave32 butterfly
            ps += __shfl_xor(ps, off, 32);
            pq += __shfl_xor(pq, off, 32);
        }
        const float mu  = ps * (1.f / CM);
        const float var = pq * (1.f / CM) - mu * mu;
        const float rs  = rsqrtf(var + LN_EPS);
#pragma unroll
        for (int u = 0; u < 8; ++u) {
            float xv = (u < 4) ? x0[u] : x1[u - 4];
            accS[u] += (xv - mu) * rs * gv[u] + bv[u];
        }
    }
#pragma unroll
    for (int u = 0; u < 8; ++u) Ssh[w][c0 + u] = accS[u];
    __syncthreads();

    // combine 8 waves: thread t owns column c = t
    float Sc = 0.f;
#pragma unroll
    for (int ww = 0; ww < 8; ++ww) Sc += Ssh[ww][t];
    Ssh[0][t] = Sc;                    // own column, no race
    __syncthreads();

    if (t < CH) {
        float y = (float)KDIM * bag[t];
        for (int c = 0; c < CM; ++c) y += Ssh[0][c] * Wag[c * CH + t];
        msum_h[j * CH + t] = (_Float16)y;
    }
}

// ---------------------------------------------------------------------------
// K4: z[i, jt*16+M, nt*16+N] = WMMA( msum_h[16x32] , Tt[i][32x16] ) + b_out*inv
// grid (64 jt, 256 i), block 256 = 8 waves, wave w -> N-tile nt = w
// one v_wmma_f32_16x16x32_f16 per output tile (K = e = 32 exactly)
// ---------------------------------------------------------------------------
__global__ void k_wmma_out(const _Float16* __restrict__ msum_h,
                           const _Float16* __restrict__ Tt,
                           const float* __restrict__ b_out,
                           float* __restrict__ out,
                           float inv_norm)
{
    const int jt   = blockIdx.x;           // 0..63
    const int i    = blockIdx.y;           // 0..255
    const int w    = threadIdx.x >> 5;     // nt = 0..7
    const int lane = threadIdx.x & 31;
    const int gh   = lane >> 4;            // lane-half g
    const int l16  = lane & 15;

    // A fragment (16x32 f16): row j = jt*16 + l16
    // halves 0..15 -> K = 16g+0..15
    const _Float16* rowA = msum_h + (jt * 16 + l16) * CH;
    v8h alo = *(const v8h*)(rowA + 16 * gh);
    v8h ahi = *(const v8h*)(rowA + 16 * gh + 8);
    v16h afrag;
#pragma unroll
    for (int u = 0; u < 8; ++u) { afrag[u] = alo[u]; afrag[8 + u] = ahi[u]; }

    // B fragment (32x16 f16): col n = w*16 + l16, halves -> K = h + 16g
    // Tt stored [i][z][e] so the 16 K-values are contiguous
    const int n = w * 16 + l16;
    const _Float16* rowB = Tt + ((size_t)i * CZ + n) * CH + 16 * gh;
    v16h bfrag = *(const v16h*)rowB;

    v8f c = {};
    v8f d = __builtin_amdgcn_wmma_f32_16x16x32_f16(
        /*neg_a=*/false, afrag, /*neg_b=*/false, bfrag,
        /*c_mod=*/(short)0, c, /*reuse_a=*/false, /*reuse_b=*/false);

    const float bos = b_out[n] * inv_norm;
    float* outp = out + ((size_t)i * L2 + jt * 16) * CZ + n;
#pragma unroll
    for (int r = 0; r < 8; ++r) {
        const int mrow = r + 8 * gh;                 // C/D layout: M = r + 8g
        outp[(size_t)mrow * CZ] = d[r] + bos;
    }
}

// ---------------------------------------------------------------------------
extern "C" void kernel_launch(void* const* d_in, const int* in_sizes, int n_in,
                              void* d_out, int out_size, void* d_ws, size_t ws_size,
                              hipStream_t stream)
{
    (void)in_sizes; (void)n_in; (void)out_size; (void)ws_size;
    const float* s     = (const float*)d_in[0];
    const float* m     = (const float*)d_in[1];
    const float* lnabg = (const float*)d_in[2];
    const float* lnabb = (const float*)d_in[3];
    const float* lnagg = (const float*)d_in[4];
    const float* lnagb = (const float*)d_in[5];
    const float* W_ab  = (const float*)d_in[6];
    const float* b_ab  = (const float*)d_in[7];
    const float* W_ag  = (const float*)d_in[8];
    const float* b_ag  = (const float*)d_in[9];
    const float* W_out = (const float*)d_in[10];
    const float* b_out = (const float*)d_in[11];
    float* out = (float*)d_out;

    char* ws = (char*)d_ws;
    float*    a_buf  = (float*)ws;                                   //  32 KB
    _Float16* Tt     = (_Float16*)(ws + 32 * 1024);                  //   2 MB
    _Float16* msum_h = (_Float16*)(ws + 32 * 1024 + 2 * 1024 * 1024);//  64 KB

    const float inv_norm = 1.0f / ((float)KDIM + 1e-3f);

    k_ln_proj_s<<<L1, 128, 0, stream>>>(s, lnabg, lnabb, W_ab, b_ab, a_buf);
    k_make_T  <<<dim3(L1, CZ / 8), dim3(32, 8), 0, stream>>>(a_buf, W_out, Tt, inv_norm);
    k_msum    <<<L2, 256, 0, stream>>>(m, lnagg, lnagb, W_ag, b_ag, msum_h);
    k_wmma_out<<<dim3(L2 / 16, L1), 256, 0, stream>>>(msum_h, Tt, b_out, out, inv_norm);
}
